// GQAMultiheadAttention_7421703487879
// MI455X (gfx1250) — matrix operand order repaired and checked by reading, not compiled, not run on hardware
//
#include <hip/hip_runtime.h>

#define DEV __device__ __forceinline__

typedef __bf16 bf16_t;
typedef __bf16 v8bf  __attribute__((ext_vector_type(8)));
typedef __bf16 v16bf __attribute__((ext_vector_type(16)));
typedef float  v8f   __attribute__((ext_vector_type(8)));

// Problem constants (B,T,C fixed by the reference)
constexpr int Bc  = 2;
constexpr int Tc  = 2048;
constexpr int Cc  = 2048;
constexpr int Hc  = 32;
constexpr int KVc = 8;
constexpr int DHc = 64;
constexpr int NQ  = Hc * DHc;   // 2048
constexpr int NKV = KVc * DHc;  // 512
constexpr int ROWS = Bc * Tc;   // 4096

DEV v8f zero8() { v8f z = {0.f,0.f,0.f,0.f,0.f,0.f,0.f,0.f}; return z; }

DEV v16bf cat16(v8bf lo, v8bf hi) {
  return __builtin_shufflevector(lo, hi, 0,1,2,3,4,5,6,7,8,9,10,11,12,13,14,15);
}

// A-matrix fragment, 16x32 bf16 (contiguous k order, B loaded to match): lane holds row (lane&15);
// vector elems 0..7  = A[row][kb .. kb+7]
// vector elems 8..15 = A[row][kb+8 .. kb+15],  kb = 16*(lane>>4) (+ k-chunk base)
DEV v16bf load_a_frag(const bf16_t* rowp, int kb) {
  v8bf lo = *(const v8bf*)(rowp + kb);
  v8bf hi = *(const v8bf*)(rowp + kb + 8);
  return cat16(lo, hi);
}

DEV v8f wmma_bf16(v16bf a, v16bf b, v8f c) {
  return __builtin_amdgcn_wmma_f32_16x16x32_bf16(false, a, false, b, (short)0, c,
                                                 false, false);
}

// ---------------------------------------------------------------------------
// Format conversion helpers
// ---------------------------------------------------------------------------
__global__ void cvt_bf16_kernel(const float* __restrict__ in,
                                bf16_t* __restrict__ out, long n) {
  long i = (long)blockIdx.x * blockDim.x + threadIdx.x;
  long stride = (long)gridDim.x * blockDim.x;
  for (; i < n; i += stride) out[i] = (bf16_t)in[i];
}

// in: rows x cols fp32 (row-major)  ->  out: cols x rows bf16 (row-major)
__global__ void transpose_bf16_kernel(const float* __restrict__ in,
                                      bf16_t* __restrict__ out,
                                      int rows, int cols) {
  long n = (long)rows * cols;
  long i = (long)blockIdx.x * blockDim.x + threadIdx.x;
  long stride = (long)gridDim.x * blockDim.x;
  for (; i < n; i += stride) {
    int r = (int)(i / cols);
    int c = (int)(i % cols);
    out[(long)c * rows + r] = (bf16_t)in[i];
  }
}

// ---------------------------------------------------------------------------
// Core 32x64 GEMM block: C(32x64,f32) += A(32xK,bf16,row-major) * Bt^T
// Bt is N x K bf16 row-major (i.e. B column n is Bt row n, contiguous in K).
// Two 16-row A tiles share each B fragment: 12 b128 loads per 8 WMMAs.
// ---------------------------------------------------------------------------
DEV void gemm32x64(const bf16_t* __restrict__ A, int lda,
                   const bf16_t* __restrict__ Bt, int ldb,
                   long row0, long col0, int K, v8f acc[2][4]) {
  const int lane = threadIdx.x & 31;
  const int half = lane >> 4;
  const bf16_t* aptr0 = A + (row0 + (lane & 15)) * (long)lda;
  const bf16_t* aptr1 = aptr0 + 16 * (long)lda;
  for (int kk = 0; kk < K; kk += 32) {
    v16bf a0 = load_a_frag(aptr0, kk + 16 * half);
    v16bf a1 = load_a_frag(aptr1, kk + 16 * half);
#pragma unroll
    for (int nt = 0; nt < 4; ++nt) {
      const bf16_t* bptr =
          Bt + (col0 + nt * 16 + (lane & 15)) * (long)ldb + kk + 16 * half;
      v16bf bf = *(const v16bf*)bptr;
      acc[0][nt] = wmma_bf16(a0, bf, acc[0][nt]);
      acc[1][nt] = wmma_bf16(a1, bf, acc[1][nt]);
    }
  }
}

// ---------------------------------------------------------------------------
// Fused QKV projection + RoPE.  32x64 output tile per wave.
//   mode 0: Q -> rope, *0.125, store row-major [ROWS][2048]
//   mode 1: K -> rope,        store [b][kv][t][64]
//   mode 2: V ->              store transposed [b][kv][64][t]
// ---------------------------------------------------------------------------
__global__ __launch_bounds__(32)
void qkv_proj_kernel(const bf16_t* __restrict__ xbf,
                     const bf16_t* __restrict__ Wt,
                     const float* __restrict__ rsin,
                     const float* __restrict__ rcos,
                     bf16_t* __restrict__ out, int mode) {
  const int lane = threadIdx.x & 31;
  const int half = lane >> 4;
  long row0 = (long)blockIdx.y * 32;
  long col0 = (long)blockIdx.x * 64;

  v8f acc[2][4] = {{zero8(), zero8(), zero8(), zero8()},
                   {zero8(), zero8(), zero8(), zero8()}};
  gemm32x64(xbf, Cc, Wt, Cc, row0, col0, Cc, acc);

#pragma unroll
  for (int rt = 0; rt < 2; ++rt) {
#pragma unroll
    for (int nt = 0; nt < 4; ++nt) {
#pragma unroll
      for (int r = 0; r < 8; ++r) {
        long row = row0 + rt * 16 + r + 8 * half;  // global token row (b*T+t)
        int t = (int)(row & (Tc - 1));             // T = 2048 (power of two)
        int bidx = (int)(row >> 11);               // row / 2048
        int col = (int)col0 + nt * 16 + (lane & 15);
        float f = acc[rt][nt][r];
        if (mode != 2) {  // RoPE for Q and K
          int d = col & 63;
          int p = d >> 1;
          float cc = rcos[t * 32 + p];
          float ss = rsin[t * 32 + p];
          float partner = __shfl_xor(f, 1);
          if ((col & 1) == 0)
            f = f * cc - partner * ss;  // o1 = x1*c - x2*s
          else
            f = partner * ss + f * cc;  // o2 = x1*s + x2*c
          if (mode == 0) f *= 0.125f;   // fold 1/sqrt(DH) into Q
        }
        if (mode == 0) {
          out[row * NQ + col] = (bf16_t)f;
        } else {
          int kvh = col >> 6;
          int d = col & 63;
          if (mode == 1) {
            out[(((long)(bidx * KVc + kvh) * Tc + t) << 6) + d] = (bf16_t)f;
          } else {
            out[((long)((bidx * KVc + kvh) * 64 + d)) * Tc + t] = (bf16_t)f;
          }
        }
      }
    }
  }
}

// ---------------------------------------------------------------------------
// Flash attention: one wave per (b, h, 16-query tile).
// ---------------------------------------------------------------------------
__global__ __launch_bounds__(32)
void attn_kernel(const bf16_t* __restrict__ Q,
                 const bf16_t* __restrict__ K,
                 const bf16_t* __restrict__ Vt,
                 const int* __restrict__ swa,
                 bf16_t* __restrict__ attn) {
  __shared__ __align__(32) bf16_t Pls[16 * 32];

  const int lane = threadIdx.x & 31;
  const int half = lane >> 4;
  const int qb = blockIdx.x * 16;
  const int h = blockIdx.y;
  const int b = blockIdx.z;
  const int kvh = h >> 2;  // GROUP = 4
  const int W = swa[0];

  // Q A-fragments (DH=64 -> two K=32 chunks), Q already scaled by 1/8
  const bf16_t* qrow =
      Q + ((long)(b * Tc) + qb + (lane & 15)) * NQ + h * DHc;
  v16bf a0 = load_a_frag(qrow, 0 + 16 * half);
  v16bf a1 = load_a_frag(qrow, 32 + 16 * half);

  const bf16_t* Kb = K + (long)(b * KVc + kvh) * Tc * DHc;   // [t][64]
  const bf16_t* Vb = Vt + (long)(b * KVc + kvh) * DHc * Tc;  // [d][t]

  v8f o[4] = {zero8(), zero8(), zero8(), zero8()};
  float m[8], l[8];
#pragma unroll
  for (int r = 0; r < 8; ++r) { m[r] = -1e30f; l[r] = 0.f; }

  int jmin = qb - W; if (jmin < 0) jmin = 0;
  const int jt0 = jmin >> 5;
  const int jt1 = (qb + 15) >> 5;

  for (int jt = jt0; jt <= jt1; ++jt) {
    const int jbase = jt * 32;

    // scores: two 16x16 N-subtiles, each accumulating two K=32 chunks
    v8f sc[2];
#pragma unroll
    for (int s = 0; s < 2; ++s) {
      const bf16_t* krow =
          Kb + (long)(jbase + s * 16 + (lane & 15)) * DHc + 16 * half;
      v16bf b0 = *(const v16bf*)krow;
      v16bf b1 = *(const v16bf*)(krow + 32);
      v8f z = zero8();
      z = wmma_bf16(a0, b0, z);
      z = wmma_bf16(a1, b1, z);
      sc[s] = z;
    }

    // online softmax over this 32-key tile
    const int jcol = lane & 15;
#pragma unroll
    for (int r = 0; r < 8; ++r) {
      int i = qb + r + 8 * half;
      int j0 = jbase + jcol;
      int j1 = j0 + 16;
      bool ok0 = (j0 <= i) && ((i - j0) <= W);
      bool ok1 = (j1 <= i) && ((i - j1) <= W);
      float v0 = ok0 ? sc[0][r] : -3.0e38f;
      float v1 = ok1 ? sc[1][r] : -3.0e38f;
      float mx = fmaxf(v0, v1);
#pragma unroll
      for (int msk = 1; msk < 16; msk <<= 1) mx = fmaxf(mx, __shfl_xor(mx, msk));
      float mn = fmaxf(m[r], mx);
      float alpha = __expf(m[r] - mn);
      float p0 = ok0 ? __expf(v0 - mn) : 0.f;
      float p1 = ok1 ? __expf(v1 - mn) : 0.f;
      float rs = p0 + p1;
#pragma unroll
      for (int msk = 1; msk < 16; msk <<= 1) rs += __shfl_xor(rs, msk);
      l[r] = l[r] * alpha + rs;
      m[r] = mn;
#pragma unroll
      for (int dt = 0; dt < 4; ++dt) o[dt][r] *= alpha;
      int M = r + 8 * half;
      Pls[M * 32 + jcol] = (bf16_t)p0;
      Pls[M * 32 + 16 + jcol] = (bf16_t)p1;
    }
    __syncthreads();

    // reload P as A-fragment (C-layout -> A-layout via LDS)
    v16bf pa = load_a_frag(Pls + (lane & 15) * 32, 16 * half);

#pragma unroll
    for (int dt = 0; dt < 4; ++dt) {
      const bf16_t* vrow =
          Vb + (long)(dt * 16 + (lane & 15)) * Tc + jbase + 16 * half;
      v16bf vb = *(const v16bf*)vrow;
      o[dt] = wmma_bf16(pa, vb, o[dt]);
    }
    __syncthreads();
  }

  // epilogue: normalize and store bf16 [b*T+t][h*64+d]
#pragma unroll
  for (int dt = 0; dt < 4; ++dt) {
#pragma unroll
    for (int r = 0; r < 8; ++r) {
      float val = (l[r] > 0.f) ? (o[dt][r] / l[r]) : 0.f;
      attn[((long)(b * Tc) + qb + r + 8 * half) * NQ + h * DHc + dt * 16 +
           (lane & 15)] = (bf16_t)val;
    }
  }
}

// ---------------------------------------------------------------------------
// Output projection: out(fp32) = attn_bf @ Wout  (via WoutT), 32x64 tiles
// ---------------------------------------------------------------------------
__global__ __launch_bounds__(32)
void out_proj_kernel(const bf16_t* __restrict__ Abf,
                     const bf16_t* __restrict__ Wt,
                     float* __restrict__ out) {
  const int lane = threadIdx.x & 31;
  const int half = lane >> 4;
  long row0 = (long)blockIdx.y * 32;
  long col0 = (long)blockIdx.x * 64;

  v8f acc[2][4] = {{zero8(), zero8(), zero8(), zero8()},
                   {zero8(), zero8(), zero8(), zero8()}};
  gemm32x64(Abf, NQ, Wt, NQ, row0, col0, NQ, acc);

#pragma unroll
  for (int rt = 0; rt < 2; ++rt) {
#pragma unroll
    for (int nt = 0; nt < 4; ++nt) {
#pragma unroll
      for (int r = 0; r < 8; ++r) {
        long row = row0 + rt * 16 + r + 8 * half;
        int col = (int)col0 + nt * 16 + (lane & 15);
        out[row * Cc + col] = acc[rt][nt][r];
      }
    }
  }
}

// ---------------------------------------------------------------------------
// Host launcher
// ---------------------------------------------------------------------------
extern "C" void kernel_launch(void* const* d_in, const int* in_sizes, int n_in,
                              void* d_out, int out_size, void* d_ws,
                              size_t ws_size, hipStream_t stream) {
  (void)in_sizes; (void)n_in; (void)out_size; (void)ws_size;
  const float* x    = (const float*)d_in[0];
  const float* Wq   = (const float*)d_in[1];
  const float* Wk   = (const float*)d_in[2];
  const float* Wv   = (const float*)d_in[3];
  const float* Wout = (const float*)d_in[4];
  const float* rsin = (const float*)d_in[5];
  const float* rcos = (const float*)d_in[6];
  const int*   swa  = (const int*)d_in[7];
  float* out = (float*)d_out;

  char* ws = (char*)d_ws;
  size_t off = 0;
  auto alloc = [&](size_t bytes) -> void* {
    void* p = ws + off;
    off = (off + bytes + 255) & ~(size_t)255;
    return p;
  };

  bf16_t* xbf = (bf16_t*)alloc((size_t)ROWS * Cc * 2);   // reused as attn out
  bf16_t* WqT = (bf16_t*)alloc((size_t)NQ * Cc * 2);
  bf16_t* WkT = (bf16_t*)alloc((size_t)NKV * Cc * 2);
  bf16_t* WvT = (bf16_t*)alloc((size_t)NKV * Cc * 2);
  bf16_t* WoT = (bf16_t*)alloc((size_t)Cc * NQ * 2);
  bf16_t* Qbf = (bf16_t*)alloc((size_t)ROWS * NQ * 2);
  bf16_t* Kbf = (bf16_t*)alloc((size_t)Bc * KVc * Tc * DHc * 2);
  bf16_t* Vtb = (bf16_t*)alloc((size_t)Bc * KVc * DHc * Tc * 2);
  bf16_t* Abf = xbf;  // attention output reuses x_bf16 storage

  // 1. convert x to bf16
  cvt_bf16_kernel<<<4096, 256, 0, stream>>>(x, xbf, (long)ROWS * Cc);
  // 2. transpose weights to [N][K] bf16
  transpose_bf16_kernel<<<4096, 256, 0, stream>>>(Wq, WqT, Cc, NQ);
  transpose_bf16_kernel<<<2048, 256, 0, stream>>>(Wk, WkT, Cc, NKV);
  transpose_bf16_kernel<<<2048, 256, 0, stream>>>(Wv, WvT, Cc, NKV);
  transpose_bf16_kernel<<<4096, 256, 0, stream>>>(Wout, WoT, NQ, Cc);

  // 3. fused QKV projections (+RoPE, +1/sqrt(DH) on Q), 32-row tiles
  qkv_proj_kernel<<<dim3(NQ / 64, ROWS / 32), 32, 0, stream>>>(
      xbf, WqT, rsin, rcos, Qbf, 0);
  qkv_proj_kernel<<<dim3(NKV / 64, ROWS / 32), 32, 0, stream>>>(
      xbf, WkT, rsin, rcos, Kbf, 1);
  qkv_proj_kernel<<<dim3(NKV / 64, ROWS / 32), 32, 0, stream>>>(
      xbf, WvT, rsin, rcos, Vtb, 2);

  // 4. flash attention with causal + sliding-window mask
  attn_kernel<<<dim3(Tc / 16, Hc, Bc), 32, 0, stream>>>(Qbf, Kbf, Vtb, swa,
                                                        Abf);

  // 5. output projection -> fp32 d_out
  out_proj_kernel<<<dim3(Cc / 64, ROWS / 32), 32, 0, stream>>>(Abf, WoT, out);
}
